// SINE_19791209300679
// MI455X (gfx1250) — matrix operand order repaired and checked by reading, not compiled, not run on hardware
//
#include <hip/hip_runtime.h>
#include <hip/hip_bf16.h>

typedef __attribute__((ext_vector_type(16))) _Float16 v16h;
typedef __attribute__((ext_vector_type(8)))  _Float16 v8h;
typedef __attribute__((ext_vector_type(2)))  _Float16 v2h;
typedef __attribute__((ext_vector_type(8)))  float    v8f;

static constexpr int LB = 128;          // batch
static constexpr int LL = 200;          // seq len
static constexpr int DD = 512;          // model dim
static constexpr int HH = 8;            // heads
static constexpr int MM = LB * LL;      // 25600 rows

__device__ __forceinline__ float wred_sum(float v) {
#pragma unroll
  for (int m = 16; m > 0; m >>= 1) v += __shfl_xor(v, m, 32);
  return v;
}
__device__ __forceinline__ float wred_max(float v) {
#pragma unroll
  for (int m = 16; m > 0; m >>= 1) v = fmaxf(v, __shfl_xor(v, m, 32));
  return v;
}

// ---------------------------------------------------------------------------
// Embedding gathers: seqs = emb[log]*sqrt(D) + pos_emb, masked; plus pos/neg/fdbk
// ---------------------------------------------------------------------------
__global__ __launch_bounds__(256) void embed_k(
    const float* __restrict__ emb, const float* __restrict__ pemb,
    const int* __restrict__ logs, const int* __restrict__ poss,
    const int* __restrict__ negs, const int* __restrict__ fbs,
    float* __restrict__ seqs, float* __restrict__ pe,
    float* __restrict__ ne, float* __restrict__ fe) {
  int r = blockIdx.x;
  int l = r % LL;
  int idl = logs[r], idp = poss[r], idn = negs[r], idf = fbs[r];
  float valid = (idl != 0) ? 1.0f : 0.0f;
  for (int d = threadIdx.x; d < DD; d += 256) {
    seqs[(size_t)r * DD + d] =
        (emb[(size_t)idl * DD + d] * 22.627416997969522f + pemb[l * DD + d]) * valid;
    pe[(size_t)r * DD + d] = emb[(size_t)idp * DD + d];
    ne[(size_t)r * DD + d] = emb[(size_t)idn * DD + d];
    fe[(size_t)r * DD + d] = emb[(size_t)idf * DD + d];
  }
}

// ---------------------------------------------------------------------------
// target[r] = argmax_k  E[k] . (posE[r] - fdbkE[r])   (one wave per row)
// ---------------------------------------------------------------------------
__global__ __launch_bounds__(256) void target_k(
    const float* __restrict__ pe, const float* __restrict__ fe,
    const float* __restrict__ E, int* __restrict__ target) {
  int wid = threadIdx.x >> 5, lane = threadIdx.x & 31;
  int r = blockIdx.x * 8 + wid;
  float acc[4] = {0.f, 0.f, 0.f, 0.f};
  for (int d = lane; d < DD; d += 32) {
    float diff = pe[(size_t)r * DD + d] - fe[(size_t)r * DD + d];
#pragma unroll
    for (int k = 0; k < 4; ++k) acc[k] += diff * E[k * DD + d];
  }
#pragma unroll
  for (int k = 0; k < 4; ++k) acc[k] = wred_sum(acc[k]);
  if (lane == 0) {
    int best = 0; float bv = acc[0];
#pragma unroll
    for (int k = 1; k < 4; ++k) if (acc[k] > bv) { bv = acc[k]; best = k; }
    target[r] = best;
  }
}

// first_i[b] = argmax_k  E[k] . emb[log[b,0]]   (one wave per b)
__global__ __launch_bounds__(256) void first_k(
    const float* __restrict__ emb, const int* __restrict__ logs,
    const float* __restrict__ E, int* __restrict__ firsti) {
  int wid = threadIdx.x >> 5, lane = threadIdx.x & 31;
  int b = blockIdx.x * 8 + wid;
  int id = logs[b * LL];
  float acc[4] = {0.f, 0.f, 0.f, 0.f};
  for (int d = lane; d < DD; d += 32) {
    float v = emb[(size_t)id * DD + d];
#pragma unroll
    for (int k = 0; k < 4; ++k) acc[k] += v * E[k * DD + d];
  }
#pragma unroll
  for (int k = 0; k < 4; ++k) acc[k] = wred_sum(acc[k]);
  if (lane == 0) {
    int best = 0; float bv = acc[0];
#pragma unroll
    for (int k = 1; k < 4; ++k) if (acc[k] > bv) { bv = acc[k]; best = k; }
    firsti[b] = best;
  }
}

__global__ __launch_bounds__(256) void inp_k(
    const int* __restrict__ target, const int* __restrict__ firsti,
    int* __restrict__ inp) {
  int r = blockIdx.x * 256 + threadIdx.x;
  if (r >= MM) return;
  int b = r / LL, l = r % LL;
  inp[r] = (l == 0) ? firsti[b] : target[r - 1];
}

// ---------------------------------------------------------------------------
// LayerNorm, one block per row (supports in==out)
// ---------------------------------------------------------------------------
__global__ __launch_bounds__(256) void ln_k(
    const float* __restrict__ in, const float* __restrict__ g,
    const float* __restrict__ bb, float* __restrict__ out, int cols) {
  int row = blockIdx.x, tid = threadIdx.x;
  const float* x = in + (size_t)row * cols;
  float s = 0.f, s2 = 0.f;
  for (int d = tid; d < cols; d += 256) { float v = x[d]; s += v; s2 += v * v; }
  s = wred_sum(s); s2 = wred_sum(s2);
  __shared__ float sh[17];
  int wid = tid >> 5, lane = tid & 31;
  if (lane == 0) { sh[wid] = s; sh[8 + wid] = s2; }
  __syncthreads();
  if (tid == 0) {
    float a = 0.f, b2 = 0.f;
#pragma unroll
    for (int i = 0; i < 8; ++i) { a += sh[i]; b2 += sh[8 + i]; }
    sh[0] = a; sh[8] = b2;
  }
  __syncthreads();
  float mean = sh[0] / cols;
  float var = sh[8] / cols - mean * mean;
  float inv = rsqrtf(var + 1e-8f);
  for (int d = tid; d < cols; d += 256)
    out[(size_t)row * cols + d] = (x[d] - mean) * inv * g[d] + bb[d];
}

// ---------------------------------------------------------------------------
// WMMA GEMM: C[M,N] = A[M,K] @ B[K,N]  (+bias +resid, relu, row-mask fused)
// 128x128 tile / workgroup, 8 waves (4x2), 2x4 WMMA accs per wave, K step 32.
// Double-buffered LDS staging: next tile's global loads (distinct registers)
// overlap the current tile's 8 WMMAs; one barrier per K-step.
// ---------------------------------------------------------------------------
__global__ __launch_bounds__(256) void gemm128(
    const float* __restrict__ A, const float* __restrict__ Bw,
    const float* __restrict__ bias, const float* __restrict__ resid,
    const int* __restrict__ rowmask, float* __restrict__ C,
    int Md, int Nd, int Kd, int relu) {
  __shared__ __align__(32) _Float16 sA[2][128][48];  // rows x K(32) padded to 48
  __shared__ __align__(32) _Float16 sB[2][128][48];  // cols(n) x K(32), transposed

  const int tid = threadIdx.x;
  const int m0 = blockIdx.y * 128;
  const int n0 = blockIdx.x * 128;
  const int wid = tid >> 5, lane = tid & 31;
  const int wm = wid & 3;          // 0..3  -> 32-row strip
  const int wn = wid >> 2;         // 0..1  -> 64-col strip
  const int half = lane >> 4, lq = lane & 15;

  const int arow = tid >> 3;            // 0..31
  const int acol = (tid & 7) * 4;       // 0..28
  const int bkr = tid >> 5;             // 0..7
  const int bnc = (tid & 31) * 4;       // 0..124

  v8f acc[2][4] = {};
  float4 ra0, ra1, ra2, ra3, rb0, rb1, rb2, rb3;

  auto load_tile = [&](int k0) {
    const float* ap = A + (size_t)(m0 + arow) * Kd + k0 + acol;
    ra0 = *(const float4*)(ap);
    ra1 = *(const float4*)(ap + (size_t)32 * Kd);
    ra2 = *(const float4*)(ap + (size_t)64 * Kd);
    ra3 = *(const float4*)(ap + (size_t)96 * Kd);
    const float* bp = Bw + (size_t)(k0 + bkr) * Nd + n0 + bnc;
    rb0 = *(const float4*)(bp);
    rb1 = *(const float4*)(bp + (size_t)8 * Nd);
    rb2 = *(const float4*)(bp + (size_t)16 * Nd);
    rb3 = *(const float4*)(bp + (size_t)24 * Nd);
  };
  auto store_tile = [&](int buf) {
    _Float16 (*cA)[48] = sA[buf];
    _Float16 (*cB)[48] = sB[buf];
    float4 rr[4] = {ra0, ra1, ra2, ra3};
#pragma unroll
    for (int p = 0; p < 4; ++p) {
      int r = arow + p * 32;
      v2h t0; t0.x = (_Float16)rr[p].x; t0.y = (_Float16)rr[p].y;
      v2h t1; t1.x = (_Float16)rr[p].z; t1.y = (_Float16)rr[p].w;
      *(v2h*)&cA[r][acol] = t0;
      *(v2h*)&cA[r][acol + 2] = t1;
    }
    float4 ss[4] = {rb0, rb1, rb2, rb3};
#pragma unroll
    for (int p = 0; p < 4; ++p) {
      int kk = bkr + p * 8;
      cB[bnc + 0][kk] = (_Float16)ss[p].x;
      cB[bnc + 1][kk] = (_Float16)ss[p].y;
      cB[bnc + 2][kk] = (_Float16)ss[p].z;
      cB[bnc + 3][kk] = (_Float16)ss[p].w;
    }
  };
  auto compute = [&](int buf) {
    _Float16 (*cA)[48] = sA[buf];
    _Float16 (*cB)[48] = sB[buf];
    union { v16h v; v8h h[2]; } af[2];
    v16h bf[4];
#pragma unroll
    for (int mt = 0; mt < 2; ++mt) {
      int r = wm * 32 + mt * 16 + lq;
      af[mt].h[0] = *(const v8h*)&cA[r][half * 16];       // K = half*16 .. +7
      af[mt].h[1] = *(const v8h*)&cA[r][half * 16 + 8];   // K = half*16+8 .. +15
    }
#pragma unroll
    for (int nt = 0; nt < 4; ++nt) {
      int n = wn * 64 + nt * 16 + lq;
      bf[nt] = *(const v16h*)&cB[n][half * 16];           // K = half*16 .. +15
    }
#pragma unroll
    for (int mt = 0; mt < 2; ++mt)
#pragma unroll
      for (int nt = 0; nt < 4; ++nt)
        acc[mt][nt] = __builtin_amdgcn_wmma_f32_16x16x32_f16(
            false, af[mt].v, false, bf[nt], (short)0, acc[mt][nt], false, false);
  };

  load_tile(0);
  store_tile(0);
  int cur = 0;
  for (int k0 = 32; k0 < Kd; k0 += 32) {
    load_tile(k0);                          // regs for tile k0 in flight
    if (k0 + 32 < Kd) {                     // gfx1250 global_prefetch, 2 tiles ahead
      __builtin_prefetch(A + (size_t)(m0 + arow) * Kd + k0 + 32 + acol, 0, 1);
      __builtin_prefetch(Bw + (size_t)(k0 + 32 + bkr) * Nd + n0 + bnc, 0, 1);
    }
    __syncthreads();                        // buf `cur` fully staged
    compute(cur);                           // WMMAs overlap outstanding loads
    store_tile(cur ^ 1);
    cur ^= 1;
  }
  __syncthreads();
  compute(cur);

  const int cm = m0 + wm * 32;
  const int cn = n0 + wn * 64;
#pragma unroll
  for (int mt = 0; mt < 2; ++mt)
#pragma unroll
    for (int nt = 0; nt < 4; ++nt) {
      int col = cn + nt * 16 + lq;
      float bv = bias ? bias[col] : 0.0f;
#pragma unroll
      for (int r = 0; r < 8; ++r) {
        int row = cm + mt * 16 + half * 8 + r;
        float v = acc[mt][nt][r] + bv;
        if (resid) v += resid[(size_t)row * Nd + col];
        if (relu) v = fmaxf(v, 0.0f);
        if (rowmask && rowmask[row] == 0) v = 0.0f;
        C[(size_t)row * Nd + col] = v;
      }
    }
}

// ---------------------------------------------------------------------------
// Attention per (b,h): K/V staged f16 in LDS; each wave owns q rows wid,wid+8..
// q row held in registers (wave-uniform). Causal softmax + intent_w fused.
// ---------------------------------------------------------------------------
__global__ __launch_bounds__(256) void attn_k(
    const float* __restrict__ qp, const float* __restrict__ kp,
    const float* __restrict__ vp, const int* __restrict__ inp,
    const int* __restrict__ target, float* __restrict__ xo) {
  __shared__ _Float16 ks[LL][68];   // padded rows to spread banks
  __shared__ _Float16 vs[LL][68];
  __shared__ float pbuf[8][LL];

  int b = blockIdx.x / HH, h = blockIdx.x % HH;
  int tid = threadIdx.x, wid = tid >> 5, lane = tid & 31;
  size_t base = ((size_t)b * LL) * DD + (size_t)h * 64;

  for (int idx = tid; idx < LL * 16; idx += 256) {
    int row = idx >> 4, c = (idx & 15) * 4;
    const float4 k4 = *(const float4*)(kp + base + (size_t)row * DD + c);
    v2h ka; ka.x = (_Float16)k4.x; ka.y = (_Float16)k4.y;
    v2h kb; kb.x = (_Float16)k4.z; kb.y = (_Float16)k4.w;
    *(v2h*)&ks[row][c] = ka;
    *(v2h*)&ks[row][c + 2] = kb;
    const float4 v4 = *(const float4*)(vp + base + (size_t)row * DD + c);
    v2h va; va.x = (_Float16)v4.x; va.y = (_Float16)v4.y;
    v2h vb; vb.x = (_Float16)v4.z; vb.y = (_Float16)v4.w;
    *(v2h*)&vs[row][c] = va;
    *(v2h*)&vs[row][c + 2] = vb;
  }
  __syncthreads();

  const int* inpb = inp + b * LL;
  const int* tgtb = target + b * LL;

  for (int q = wid; q < LL; q += 8) {   // equal trip count for all 8 waves
    const float* qrow = qp + base + (size_t)q * DD;
    float qreg[64];
#pragma unroll
    for (int d = 0; d < 64; ++d) qreg[d] = qrow[d];   // wave-uniform row

    int tq = tgtb[q];
    float pk[7];
    int nk = 0;
    float mx = -1e30f;
    for (int kk = lane; kk < LL; kk += 32) {
      float dot = 0.f;
#pragma unroll
      for (int d = 0; d < 64; ++d) dot += qreg[d] * (float)ks[kk][d];
      float sc = dot * 0.125f;               // 1/sqrt(64)
      if (kk > q) sc = -1e30f;               // causal mask
      pk[nk++] = sc;
      mx = fmaxf(mx, sc);
    }
    mx = wred_max(mx);
    float sum = 0.f;
    {
      int kk = lane;
      for (int i = 0; i < nk; ++i, kk += 32) {
        float e = (pk[i] <= -1e29f) ? 0.0f : __expf(pk[i] - mx);
        pk[i] = e; sum += e;
      }
    }
    sum = wred_sum(sum);
    float inv = 1.0f / sum;
    {
      int kk = lane;
      for (int i = 0; i < nk; ++i, kk += 32) {
        float w = pk[i] * inv;
        if (kk <= q && inpb[kk] == tq) w *= 0.5f;   // intent_w
        pbuf[wid][kk] = w;
      }
    }
    __syncthreads();

    float o0 = 0.f, o1 = 0.f;
    for (int kk = 0; kk <= q; ++kk) {
      float w = pbuf[wid][kk];
      o0 += w * (float)vs[kk][lane];
      o1 += w * (float)vs[kk][lane + 32];
    }
    xo[base + (size_t)q * DD + lane]      = o0;
    xo[base + (size_t)q * DD + lane + 32] = o1;
    __syncthreads();
  }
}

// ---------------------------------------------------------------------------
// Final: LN + interest gating + weighted logits. One wave per (b,l) row.
// score softmax is softmax_k(lf . ia_w1): ia_b and e.ia_w2 are constant in k.
// ---------------------------------------------------------------------------
__global__ __launch_bounds__(256) void final_k(
    const float* __restrict__ seqs, const float* __restrict__ g,
    const float* __restrict__ bb, const float* __restrict__ E,
    const float* __restrict__ pe, const float* __restrict__ ne,
    const float* __restrict__ fe, const float* __restrict__ w1,
    float* __restrict__ out) {
  int wid = threadIdx.x >> 5, lane = threadIdx.x & 31;
  int r = blockIdx.x * 8 + wid;
  const float* x = seqs + (size_t)r * DD;
  float xv[16];
  float s = 0.f, s2 = 0.f;
#pragma unroll
  for (int j = 0; j < 16; ++j) {
    float v = x[lane + 32 * j];
    xv[j] = v; s += v; s2 += v * v;
  }
  s = wred_sum(s); s2 = wred_sum(s2);
  float mean = s * (1.0f / DD);
  float var = s2 * (1.0f / DD) - mean * mean;
  float inv = rsqrtf(var + 1e-8f);

  float dp[4] = {0, 0, 0, 0}, dn[4] = {0, 0, 0, 0};
  float df[4] = {0, 0, 0, 0}, dw[4] = {0, 0, 0, 0};
#pragma unroll
  for (int j = 0; j < 16; ++j) {
    int d = lane + 32 * j;
    float t = (xv[j] - mean) * inv * g[d] + bb[d];
    float pv = pe[(size_t)r * DD + d];
    float nv = ne[(size_t)r * DD + d];
    float fv = fe[(size_t)r * DD + d];
    float wv = w1[d];
#pragma unroll
    for (int k = 0; k < 4; ++k) {
      float ek = E[k * DD + d];
      float lf = t + t / (1.0f + __expf(-t * ek));   // t + sigmoid(t*ek)*t
      dp[k] += lf * pv; dn[k] += lf * nv; df[k] += lf * fv; dw[k] += lf * wv;
    }
  }
#pragma unroll
  for (int k = 0; k < 4; ++k) {
    dp[k] = wred_sum(dp[k]); dn[k] = wred_sum(dn[k]);
    df[k] = wred_sum(df[k]); dw[k] = wred_sum(dw[k]);
  }
  if (lane == 0) {
    float mb = fmaxf(fmaxf(dw[0], dw[1]), fmaxf(dw[2], dw[3]));
    float sk[4], se = 0.f;
#pragma unroll
    for (int k = 0; k < 4; ++k) { sk[k] = __expf(dw[k] - mb); se += sk[k]; }
    float p = 0.f, n = 0.f, f = 0.f;
#pragma unroll
    for (int k = 0; k < 4; ++k) {
      float w = sk[k] / se;
      p += w * dp[k]; n += w * dn[k]; f += w * df[k];
    }
    out[r] = p;
    out[MM + r] = n;
    out[2 * MM + r] = p - f;
  }
}

// ---------------------------------------------------------------------------
extern "C" void kernel_launch(void* const* d_in, const int* in_sizes, int n_in,
                              void* d_out, int out_size, void* d_ws, size_t ws_size,
                              hipStream_t stream) {
  (void)in_sizes; (void)n_in; (void)out_size; (void)ws_size;
  const int* logs = (const int*)d_in[0];
  const int* poss = (const int*)d_in[1];
  const int* negs = (const int*)d_in[2];
  const int* fbs  = (const int*)d_in[3];
  const float* item_emb = (const float*)d_in[4];
  const float* pos_emb  = (const float*)d_in[5];
  const float* E        = (const float*)d_in[6];
  const float* ia_w1    = (const float*)d_in[7];
  // d_in[8] = ia_w2, d_in[9] = ia_b : constant across softmax axis -> cancel
  const float* lng = (const float*)d_in[10];
  const float* lnb = (const float*)d_in[11];

  float* ws = (float*)d_ws;
  const size_t SZ = (size_t)MM * DD;
  float* seqs = ws;
  float* pe = ws + SZ;
  float* ne = ws + 2 * SZ;
  float* fe = ws + 3 * SZ;
  float* Q  = ws + 4 * SZ;
  float* qp = ws + 5 * SZ;
  float* kp = ws + 6 * SZ;
  float* vp = ws + 7 * SZ;
  float* xo = qp;   // alias: attn reads qp row q, then writes x row q (disjoint per wave)
  float* hb = kp;   // alias: FFN hidden reuses kp after attention
  int* target = (int*)(ws + 8 * SZ);
  int* inp    = target + MM;
  int* firsti = inp + MM;
  float* out = (float*)d_out;

  embed_k<<<MM, 256, 0, stream>>>(item_emb, pos_emb, logs, poss, negs, fbs,
                                  seqs, pe, ne, fe);
  target_k<<<MM / 8, 256, 0, stream>>>(pe, fe, E, target);
  first_k<<<LB / 8, 256, 0, stream>>>(item_emb, logs, E, firsti);
  inp_k<<<MM / 256, 256, 0, stream>>>(target, firsti, inp);

  dim3 gg(DD / 128, MM / 128);   // (4, 200)
  for (int blk = 0; blk < 2; ++blk) {
    const float* ln1g = (const float*)d_in[12 + blk * 12 + 0];
    const float* ln1b = (const float*)d_in[12 + blk * 12 + 1];
    const float* Wq   = (const float*)d_in[12 + blk * 12 + 2];
    const float* Wk   = (const float*)d_in[12 + blk * 12 + 3];
    const float* Wv   = (const float*)d_in[12 + blk * 12 + 4];
    const float* Wo   = (const float*)d_in[12 + blk * 12 + 5];
    const float* ln2g = (const float*)d_in[12 + blk * 12 + 6];
    const float* ln2b = (const float*)d_in[12 + blk * 12 + 7];
    const float* W1   = (const float*)d_in[12 + blk * 12 + 8];
    const float* b1   = (const float*)d_in[12 + blk * 12 + 9];
    const float* W2   = (const float*)d_in[12 + blk * 12 + 10];
    const float* b2   = (const float*)d_in[12 + blk * 12 + 11];

    ln_k<<<MM, 256, 0, stream>>>(seqs, ln1g, ln1b, Q, DD);
    gemm128<<<gg, 256, 0, stream>>>(Q,    Wq, nullptr, nullptr, nullptr, qp, MM, DD, DD, 0);
    gemm128<<<gg, 256, 0, stream>>>(seqs, Wk, nullptr, nullptr, nullptr, kp, MM, DD, DD, 0);
    gemm128<<<gg, 256, 0, stream>>>(seqs, Wv, nullptr, nullptr, nullptr, vp, MM, DD, DD, 0);
    attn_k<<<LB * HH, 256, 0, stream>>>(qp, kp, vp, inp, target, xo);
    gemm128<<<gg, 256, 0, stream>>>(xo, Wo, nullptr, Q, nullptr, seqs, MM, DD, DD, 0);
    ln_k<<<MM, 256, 0, stream>>>(seqs, ln2g, ln2b, seqs, DD);
    gemm128<<<gg, 256, 0, stream>>>(seqs, W1, b1, nullptr, nullptr, hb, MM, DD, DD, 1);
    gemm128<<<gg, 256, 0, stream>>>(hb, W2, b2, seqs, logs, seqs, MM, DD, DD, 0);
  }

  final_k<<<MM / 8, 256, 0, stream>>>(seqs, lng, lnb, E, pe, ne, fe, ia_w1, out);
}
